// Attention_12317966205037
// MI455X (gfx1250) — matrix operand order repaired and checked by reading, not compiled, not run on hardware
//
#include <hip/hip_runtime.h>

#define DIM       1024
#define HEADS     16
#define HEAD_DIM  64
#define NTOK      2048
#define BATCH     2
#define ROWS      (BATCH * NTOK)   // 4096

typedef __attribute__((ext_vector_type(16))) _Float16 v16h;
typedef __attribute__((ext_vector_type(8)))  _Float16 v8h;
typedef __attribute__((ext_vector_type(4)))  _Float16 v4h;
typedef __attribute__((ext_vector_type(8)))  float    v8f;
typedef __attribute__((ext_vector_type(4)))  float    v4f;

static __device__ __forceinline__ v8f wmma16x16x32(v16h a, v16h b, v8f c) {
  return __builtin_amdgcn_wmma_f32_16x16x32_f16(
      /*neg_a=*/false, a, /*neg_b=*/false, b,
      /*c_mod=*/(short)0, c, /*reuse_a=*/false, /*reuse_b=*/false);
}

// A-fragment (16x32 f16): lane L holds row m=L%16.
// halves 0..15 -> K = base + (L/16)*16 + h
static __device__ __forceinline__ v16h load_afrag(const _Float16* p) {
  v16h f = *(const v16h*)p;
  return f;
}

// ---------------------------------------------------------------------------
// fp32 -> fp16 conversion kernels
// ---------------------------------------------------------------------------
__global__ void cvt_f16(const float* __restrict__ s, _Float16* __restrict__ d, int n4) {
  int i = blockIdx.x * blockDim.x + threadIdx.x;   // n4 = n/4
  if (i < n4) {
    v4f x = *(const v4f*)(s + 4 * (size_t)i);
    v4h y;
#pragma unroll
    for (int t = 0; t < 4; ++t) y[t] = (_Float16)x[t];
    *(v4h*)(d + 4 * (size_t)i) = y;
  }
}

__global__ void cvt_f16_T(const float* __restrict__ s, _Float16* __restrict__ d,
                          int rows, int cols) {
  int i = blockIdx.x * blockDim.x + threadIdx.x;
  if (i < rows * cols) {
    int r = i / cols, c = i % cols;
    d[(size_t)c * rows + r] = (_Float16)s[i];
  }
}

// ---------------------------------------------------------------------------
// GEMM: C[4096 x N] = A[4096 x 1024](f16 rm) * B[1024 x N]; Bt[n*1024+k].
// Block = 128 thr (4 waves); block tile 128x64; wave tile 32x64.
// Ping-pong double-buffered over kt (unroll-by-2 => no register rotation).
// mode 0: QKV scatter (Q scaled, K, V^T, f16). mode 1: proj (+bias, f32 out).
// ---------------------------------------------------------------------------
__global__ __launch_bounds__(128) void gemm_f16_wmma(
    const _Float16* __restrict__ A,
    const _Float16* __restrict__ Bt,
    int mode,
    _Float16* __restrict__ qb, _Float16* __restrict__ kb, _Float16* __restrict__ vtb,
    const float* __restrict__ bias, float* __restrict__ outp)
{
  const int lane = threadIdx.x & 31;
  const int wave = threadIdx.x >> 5;
  const int m    = lane & 15;
  const int g    = lane >> 4;
  const int colb = blockIdx.x * 64;
  const int row0 = blockIdx.y * 128 + wave * 32;

  const _Float16* arow0 = A + (size_t)(row0 + m) * DIM;
  const _Float16* arow1 = A + (size_t)(row0 + 16 + m) * DIM;
  const _Float16* brow[4];
#pragma unroll
  for (int j = 0; j < 4; ++j)
    brow[j] = Bt + (size_t)(colb + j * 16 + m) * DIM + g * 16;

  v8f acc[2][4] = {};

  v16h af0[2], af1[2], bf[2][4];
  // prologue: buffer 0 <- kt = 0
  af0[0] = load_afrag(arow0 + g * 16);
  af1[0] = load_afrag(arow1 + g * 16);
#pragma unroll
  for (int j = 0; j < 4; ++j) bf[0][j] = *(const v16h*)(brow[j]);

#pragma unroll 1
  for (int kt = 0; kt < DIM / 32; kt += 2) {
    // L2/WGP$ prefetch a few k-tiles ahead
    const int ktp = (kt + 6) & (DIM / 32 - 1);
    __builtin_prefetch(arow0 + ktp * 32, 0, 3);
    __builtin_prefetch(brow[0] + ktp * 32, 0, 3);

    // buffer 1 <- kt+1 (kt even, so kt+1 <= 31 always)
    af0[1] = load_afrag(arow0 + (kt + 1) * 32 + g * 16);
    af1[1] = load_afrag(arow1 + (kt + 1) * 32 + g * 16);
#pragma unroll
    for (int j = 0; j < 4; ++j) bf[1][j] = *(const v16h*)(brow[j] + (kt + 1) * 32);

    // compute on buffer 0 (kt)
#pragma unroll
    for (int j = 0; j < 4; ++j) acc[0][j] = wmma16x16x32(af0[0], bf[0][j], acc[0][j]);
#pragma unroll
    for (int j = 0; j < 4; ++j) acc[1][j] = wmma16x16x32(af1[0], bf[0][j], acc[1][j]);

    // buffer 0 <- kt+2 (wraps harmlessly on the final iteration)
    const int kn = (kt + 2) & (DIM / 32 - 1);
    af0[0] = load_afrag(arow0 + kn * 32 + g * 16);
    af1[0] = load_afrag(arow1 + kn * 32 + g * 16);
#pragma unroll
    for (int j = 0; j < 4; ++j) bf[0][j] = *(const v16h*)(brow[j] + kn * 32);

    // compute on buffer 1 (kt+1)
#pragma unroll
    for (int j = 0; j < 4; ++j) acc[0][j] = wmma16x16x32(af0[1], bf[1][j], acc[0][j]);
#pragma unroll
    for (int j = 0; j < 4; ++j) acc[1][j] = wmma16x16x32(af1[1], bf[1][j], acc[1][j]);
  }

#pragma unroll
  for (int ss = 0; ss < 2; ++ss) {
#pragma unroll
    for (int j = 0; j < 4; ++j) {
#pragma unroll
      for (int r = 0; r < 8; ++r) {
        const int rr = row0 + ss * 16 + r + 8 * g;
        const int cc = colb + j * 16 + m;
        const float v = acc[ss][j][r];
        if (mode == 0) {
          const int b = rr >> 11, n = rr & 2047;
          const int three = cc >> 10, h = (cc >> 6) & 15, e = cc & 63;
          const size_t bh = (size_t)(b * HEADS + h);
          if (three == 0)
            qb[(bh * NTOK + n) * HEAD_DIM + e] = (_Float16)(v * 0.125f);
          else if (three == 1)
            kb[(bh * NTOK + n) * HEAD_DIM + e] = (_Float16)v;
          else
            vtb[(bh * HEAD_DIM + e) * NTOK + n] = (_Float16)v;
        } else {
          outp[(size_t)rr * DIM + cc] = v + bias[cc];
        }
      }
    }
  }
}

// ---------------------------------------------------------------------------
// Flash attention: block = (b,h, 64 query rows); 128 threads (4 waves).
// V fragment loads retire under the softmax VALU work; next K/V tiles are
// prefetched into WGP$/L2.
// ---------------------------------------------------------------------------
__global__ __launch_bounds__(128) void attention_wmma(
    const _Float16* __restrict__ qb, const _Float16* __restrict__ kb,
    const _Float16* __restrict__ vtb, _Float16* __restrict__ attn_h)
{
  __shared__ __align__(32) _Float16 lds_p[4][16 * 64];  // per-wave P strip

  const int lane = threadIdx.x & 31;
  const int wave = threadIdx.x >> 5;
  const int m    = lane & 15;
  const int g    = lane >> 4;
  const int qt   = blockIdx.x & 31;
  const int bh   = blockIdx.x >> 5;
  const int b    = bh >> 4;
  const int h    = bh & 15;

  const _Float16* qp = qb  + (size_t)bh * NTOK * HEAD_DIM;
  const _Float16* kp = kb  + (size_t)bh * NTOK * HEAD_DIM;
  const _Float16* vp = vtb + (size_t)bh * HEAD_DIM * NTOK;  // [hd][n]

  const int q0 = qt * 64 + wave * 16;

  v16h qf[2];
#pragma unroll
  for (int kk = 0; kk < 2; ++kk)
    qf[kk] = load_afrag(qp + (size_t)(q0 + m) * HEAD_DIM + kk * 32 + g * 16);

  v8f oacc[4] = {};
  float mrow[8], lrow[8];
#pragma unroll
  for (int r = 0; r < 8; ++r) { mrow[r] = -1e30f; lrow[r] = 0.f; }

  _Float16* pl = &lds_p[wave][0];

#pragma unroll 1
  for (int kt = 0; kt < NTOK / 64; ++kt) {
    // ---- S = Q * K^T (Q pre-scaled by 1/sqrt(hd)) ----
    v8f s[4] = {};
#pragma unroll
    for (int kk = 0; kk < 2; ++kk)
#pragma unroll
      for (int j = 0; j < 4; ++j) {
        v16h kf = *(const v16h*)(kp + (size_t)(kt * 64 + j * 16 + m) * HEAD_DIM
                                    + kk * 32 + g * 16);
        s[j] = wmma16x16x32(qf[kk], kf, s[j]);
      }

    // ---- issue V fragment loads now; they retire under the softmax ----
    v16h vf[2][4];
#pragma unroll
    for (int kk = 0; kk < 2; ++kk)
#pragma unroll
      for (int j = 0; j < 4; ++j)
        vf[kk][j] = *(const v16h*)(vp + (size_t)(j * 16 + m) * NTOK
                                      + kt * 64 + kk * 32 + g * 16);

    // WGP$/L2 prefetch of next tile's K rows and V columns
    if (kt + 1 < NTOK / 64) {
      __builtin_prefetch(kp + (size_t)(kt + 1) * 64 * HEAD_DIM + lane * 128, 0, 3);
      __builtin_prefetch(vp + (size_t)lane * NTOK + (kt + 1) * 64, 0, 3);
      __builtin_prefetch(vp + (size_t)(lane + 32) * NTOK + (kt + 1) * 64, 0, 3);
    }

    // ---- online softmax ----
#pragma unroll
    for (int r = 0; r < 8; ++r) {
      float mx = fmaxf(fmaxf(s[0][r], s[1][r]), fmaxf(s[2][r], s[3][r]));
#pragma unroll
      for (int off = 1; off < 16; off <<= 1) mx = fmaxf(mx, __shfl_xor(mx, off, 32));
      const float mnew  = fmaxf(mrow[r], mx);
      const float alpha = __expf(mrow[r] - mnew);
      float rsum = 0.f;
#pragma unroll
      for (int j = 0; j < 4; ++j) {
        float p = __expf(s[j][r] - mnew);
        s[j][r] = p;
        rsum += p;
      }
#pragma unroll
      for (int off = 1; off < 16; off <<= 1) rsum += __shfl_xor(rsum, off, 32);
      lrow[r] = lrow[r] * alpha + rsum;
      mrow[r] = mnew;
#pragma unroll
      for (int j = 0; j < 4; ++j) oacc[j][r] *= alpha;
    }

    // ---- C-layout -> A-layout relayout of P via this wave's LDS strip ----
    __builtin_amdgcn_wave_barrier();
#pragma unroll
    for (int j = 0; j < 4; ++j)
#pragma unroll
      for (int r = 0; r < 8; ++r)
        pl[(r + 8 * g) * 64 + j * 16 + m] = (_Float16)s[j][r];
    __builtin_amdgcn_wave_barrier();   // DS ops in-order per wave

    // ---- O += P * V ----
#pragma unroll
    for (int kk = 0; kk < 2; ++kk) {
      v16h pf = load_afrag(pl + m * 64 + kk * 32 + g * 16);
#pragma unroll
      for (int j = 0; j < 4; ++j)
        oacc[j] = wmma16x16x32(pf, vf[kk][j], oacc[j]);
    }
    __builtin_amdgcn_wave_barrier();
  }

  // ---- normalize and store to [B,N,H*64] f16 for proj GEMM ----
#pragma unroll
  for (int j = 0; j < 4; ++j)
#pragma unroll
    for (int r = 0; r < 8; ++r) {
      const int q   = q0 + r + 8 * g;
      const float v = oacc[j][r] / lrow[r];
      attn_h[((size_t)(b * NTOK + q)) * DIM + h * 64 + j * 16 + m] = (_Float16)v;
    }
}

// ---------------------------------------------------------------------------
extern "C" void kernel_launch(void* const* d_in, const int* in_sizes, int n_in,
                              void* d_out, int out_size, void* d_ws, size_t ws_size,
                              hipStream_t stream) {
  const float* x      = (const float*)d_in[0];
  const float* w_qkv  = (const float*)d_in[1];
  const float* w_proj = (const float*)d_in[2];
  const float* b_proj = (const float*)d_in[3];
  float* out = (float*)d_out;

  char* ws = (char*)d_ws;
  size_t off = 0;
  _Float16* x_h    = (_Float16*)(ws + off); off += (size_t)ROWS * DIM * 2;
  _Float16* wqkvT  = (_Float16*)(ws + off); off += (size_t)3 * DIM * DIM * 2;
  _Float16* wprojT = (_Float16*)(ws + off); off += (size_t)DIM * DIM * 2;
  _Float16* qb     = (_Float16*)(ws + off); off += (size_t)ROWS * DIM * 2;
  _Float16* kb     = (_Float16*)(ws + off); off += (size_t)ROWS * DIM * 2;
  _Float16* vtb    = (_Float16*)(ws + off); off += (size_t)ROWS * DIM * 2;
  _Float16* attn_h = (_Float16*)(ws + off); off += (size_t)ROWS * DIM * 2;
  (void)ws_size; (void)in_sizes; (void)n_in; (void)out_size;

  cvt_f16  <<<(ROWS * DIM / 4) / 256, 256, 0, stream>>>(x, x_h, ROWS * DIM / 4);
  cvt_f16_T<<<(3 * DIM * DIM) / 256, 256, 0, stream>>>(w_qkv, wqkvT, DIM, 3 * DIM);
  cvt_f16_T<<<(DIM * DIM) / 256, 256, 0, stream>>>(w_proj, wprojT, DIM, DIM);

  // QKV GEMM: [4096,1024] x [1024,3072] -> Q(scaled)/K/V^T
  gemm_f16_wmma<<<dim3(3 * DIM / 64, ROWS / 128), 128, 0, stream>>>(
      x_h, wqkvT, 0, qb, kb, vtb, nullptr, nullptr);

  // flash attention per (b, h, 64-query tile)
  attention_wmma<<<dim3(BATCH * HEADS * (NTOK / 64)), 128, 0, stream>>>(
      qb, kb, vtb, attn_h);

  // proj GEMM: [4096,1024] x [1024,1024] + bias -> f32 out
  gemm_f16_wmma<<<dim3(DIM / 64, ROWS / 128), 128, 0, stream>>>(
      attn_h, wprojT, 1, nullptr, nullptr, nullptr, b_proj, out);
}
